// Flux2ParallelSelfAttention_75041668596520
// MI455X (gfx1250) — matrix operand order repaired and checked by reading, not compiled, not run on hardware
//
#include <hip/hip_runtime.h>
#include <stdint.h>

// ---------------------------------------------------------------------------
// Types for CDNA5 WMMA (wave32, V_WMMA_F32_16X16X32_BF16)
// ---------------------------------------------------------------------------
typedef __attribute__((ext_vector_type(16))) __bf16 v16bf;
typedef __attribute__((ext_vector_type(8)))  float  v8f;
typedef __attribute__((ext_vector_type(4)))  unsigned int v4u;
typedef __attribute__((ext_vector_type(4)))  int v4i;

union FragB { v4u q[2]; v16bf v; };   // 32 bytes = 8 VGPRs (A or B operand)

#define HEADS     24
#define DIM_HEAD  128
#define QUERY_DIM 3072
#define INNER     3072
#define MLP_HID   12288
#define SEQ       2048
#define PROJ_OUT  33792   /* 3*INNER + 2*MLP_HID */
#define CAT_DIM   15360   /* INNER + MLP_HID     */

// Async global->LDS staging path (gfx1250 GLOBAL_LOAD_ASYNC_TO_LDS_B128,
// tracked with ASYNCcnt). Signature (from compiler diagnostic):
//   (v4i addrspace(1)* gaddr, v4i addrspace(3)* lds, imm int, imm int)
#if defined(__AMDGCN__) && defined(__gfx1250__) && \
    __has_builtin(__builtin_amdgcn_global_load_async_to_lds_b128) && \
    __has_builtin(__builtin_amdgcn_s_wait_asynccnt)
#define USE_ASYNC_LDS 1
typedef __attribute__((address_space(1))) v4i* gas_v4i_ptr;
typedef __attribute__((address_space(3))) v4i* las_v4i_ptr;
#define ASYNC_B128(gp, lp) \
  __builtin_amdgcn_global_load_async_to_lds_b128( \
      (gas_v4i_ptr)(v4i*)(gp), (las_v4i_ptr)(v4i*)(lp), 0, 0)
#else
#define USE_ASYNC_LDS 0
#endif

// f32 -> bf16 round-to-nearest-even (raw u16 storage, header-independent)
__device__ __forceinline__ unsigned short f2bf(float f) {
  union { float f; unsigned u; } v; v.f = f;
  unsigned r = v.u + 0x7FFFu + ((v.u >> 16) & 1u);
  return (unsigned short)(r >> 16);
}

__device__ __forceinline__ float grpMax16(float x) {
  #pragma unroll
  for (int m = 8; m >= 1; m >>= 1) x = fmaxf(x, __shfl_xor(x, m, 32));
  return x;
}
__device__ __forceinline__ float grpSum16(float x) {
  #pragma unroll
  for (int m = 8; m >= 1; m >>= 1) x += __shfl_xor(x, m, 32);
  return x;
}
__device__ __forceinline__ float waveSum32(float x) {
  #pragma unroll
  for (int m = 16; m >= 1; m >>= 1) x += __shfl_xor(x, m, 32);
  return x;
}

// ---------------------------------------------------------------------------
// Elementwise f32 -> bf16 cast
// ---------------------------------------------------------------------------
__global__ void cast_f32_bf16(const float* __restrict__ src,
                              unsigned short* __restrict__ dst, long n) {
  long i = (long)blockIdx.x * blockDim.x + threadIdx.x;
  long stride = (long)gridDim.x * blockDim.x;
  for (; i < n; i += stride) dst[i] = f2bf(src[i]);
}

// ---------------------------------------------------------------------------
// NT GEMM: C[M,N] = A[M,K] * B[N,K]^T (+bias).  A,B bf16(raw u16), C f32.
// Block tile 128x128, BK=32, 8 waves (4 in M x 2 in N), wave tile 32x64.
// Double-buffered LDS; staging via async loads to LDS when available.
// ---------------------------------------------------------------------------
__global__ __launch_bounds__(256) void gemm_bf16_nt(
    const unsigned short* __restrict__ A, const unsigned short* __restrict__ B,
    float* __restrict__ C, const float* __restrict__ bias,
    int M, int N, int K) {
  __shared__ __align__(16) unsigned short As[2][128 * 32];
  __shared__ __align__(16) unsigned short Bs[2][128 * 32];

  const int tid  = threadIdx.x;
  const int lane = tid & 31, wid = tid >> 5;
  const int h    = lane >> 4, ln = lane & 15;           // half / lane-in-half
  const int tileM = blockIdx.y * 128, tileN = blockIdx.x * 128;
  const int waveM = (wid & 3) * 32,   waveN = (wid >> 2) * 64;

  v8f acc[2][4] = {};

  const int nTiles = K / 32;

#if USE_ASYNC_LDS
  // Each thread owns 4 async b128 transfers per tile (2 for A, 2 for B).
  {
    #pragma unroll
    for (int t = 0; t < 2; ++t) {
      int idx = tid + t * 256;          // 0..511 ; idx = row*4 + chunk
      int row = idx >> 2, cc = idx & 3;
      ASYNC_B128(A + (size_t)(tileM + row) * K + cc * 8, &As[0][idx * 8]);
      ASYNC_B128(B + (size_t)(tileN + row) * K + cc * 8, &Bs[0][idx * 8]);
    }
  }
  for (int it = 0; it < nTiles; ++it) {
    const int buf = it & 1;
    const bool hasNext = (it + 1) < nTiles;
    if (hasNext) {
      const int kk = (it + 1) * 32;
      #pragma unroll
      for (int t = 0; t < 2; ++t) {
        int idx = tid + t * 256;
        int row = idx >> 2, cc = idx & 3;
        ASYNC_B128(A + (size_t)(tileM + row) * K + kk + cc * 8, &As[buf ^ 1][idx * 8]);
        ASYNC_B128(B + (size_t)(tileN + row) * K + kk + cc * 8, &Bs[buf ^ 1][idx * 8]);
      }
      // newest 4 may remain outstanding; everything older (current tile) done
      __builtin_amdgcn_s_wait_asynccnt(4);
    } else {
      __builtin_amdgcn_s_wait_asynccnt(0);
    }
    __syncthreads();                    // current tile visible to all waves

    const v4u* As4 = (const v4u*)As[buf];
    const v4u* Bs4 = (const v4u*)Bs[buf];
    FragB af[2];
    #pragma unroll
    for (int mf = 0; mf < 2; ++mf) {
      int r = waveM + mf * 16 + ln;
      af[mf].q[0] = As4[r * 4 + h];
      af[mf].q[1] = As4[r * 4 + 2 + h];
    }
    FragB bf_[4];
    #pragma unroll
    for (int nf = 0; nf < 4; ++nf) {
      int r = waveN + nf * 16 + ln;
      bf_[nf].q[0] = Bs4[r * 4 + h];
      bf_[nf].q[1] = Bs4[r * 4 + 2 + h];
    }
    #pragma unroll
    for (int mf = 0; mf < 2; ++mf)
      #pragma unroll
      for (int nf = 0; nf < 4; ++nf)
        acc[mf][nf] = __builtin_amdgcn_wmma_f32_16x16x32_bf16(
            false, af[mf].v, false, bf_[nf].v, (short)0, acc[mf][nf],
            false, false);
    __syncthreads();                    // all waves done reading buf before the
                                        // (it+2) iteration overwrites it
  }
#else
  for (int it = 0; it < nTiles; ++it) {
    const int kk = it * 32;
    #pragma unroll
    for (int t = 0; t < 2; ++t) {
      int idx = tid + t * 256;          // 0..511 ; idx = row*4 + chunk
      int row = idx >> 2, cc = idx & 3;
      ((v4u*)As[0])[idx] = *(const v4u*)(A + (size_t)(tileM + row) * K + kk + cc * 8);
      ((v4u*)Bs[0])[idx] = *(const v4u*)(B + (size_t)(tileN + row) * K + kk + cc * 8);
    }
    __syncthreads();
    const v4u* As4 = (const v4u*)As[0];
    const v4u* Bs4 = (const v4u*)Bs[0];
    FragB af[2];
    #pragma unroll
    for (int mf = 0; mf < 2; ++mf) {
      int r = waveM + mf * 16 + ln;
      af[mf].q[0] = As4[r * 4 + h];
      af[mf].q[1] = As4[r * 4 + 2 + h];
    }
    FragB bf_[4];
    #pragma unroll
    for (int nf = 0; nf < 4; ++nf) {
      int r = waveN + nf * 16 + ln;
      bf_[nf].q[0] = Bs4[r * 4 + h];
      bf_[nf].q[1] = Bs4[r * 4 + 2 + h];
    }
    #pragma unroll
    for (int mf = 0; mf < 2; ++mf)
      #pragma unroll
      for (int nf = 0; nf < 4; ++nf)
        acc[mf][nf] = __builtin_amdgcn_wmma_f32_16x16x32_bf16(
            false, af[mf].v, false, bf_[nf].v, (short)0, acc[mf][nf],
            false, false);
    __syncthreads();
  }
#endif

  // Epilogue per ISA D layout: lanes 0-15 -> M=r, lanes 16-31 -> M=8+r; N=ln
  #pragma unroll
  for (int mf = 0; mf < 2; ++mf)
    #pragma unroll
    for (int nf = 0; nf < 4; ++nf)
      #pragma unroll
      for (int r = 0; r < 8; ++r) {
        int row = tileM + waveM + mf * 16 + r + 8 * h;
        int col = tileN + waveN + nf * 16 + ln;
        float v = acc[mf][nf][r];
        if (bias) v += bias[col];
        C[(size_t)row * N + col] = v;
      }
}

// ---------------------------------------------------------------------------
// Fused per-head RMSNorm + RoPE on q,k; bf16 convert; V transposed to
// [head][dim][S] so attention PV B-fragments are contiguous loads.
// One wave per (s, head); each lane owns 4 consecutive dims (2 RoPE pairs).
// ---------------------------------------------------------------------------
__global__ __launch_bounds__(32) void qkv_norm_rope(
    const float* __restrict__ hbuf, const float* __restrict__ cosb,
    const float* __restrict__ sinb, const float* __restrict__ wq,
    const float* __restrict__ wk,
    unsigned short* __restrict__ qh, unsigned short* __restrict__ kh,
    unsigned short* __restrict__ vT) {
  const int spos = blockIdx.x, head = blockIdx.y;
  const int d0 = threadIdx.x * 4;
  const float* row = hbuf + (size_t)spos * PROJ_OUT;

  float q[4], k[4], v[4];
  #pragma unroll
  for (int i = 0; i < 4; ++i) {
    q[i] = row[head * DIM_HEAD + d0 + i];
    k[i] = row[INNER + head * DIM_HEAD + d0 + i];
    v[i] = row[2 * INNER + head * DIM_HEAD + d0 + i];
  }
  float sq = q[0]*q[0] + q[1]*q[1] + q[2]*q[2] + q[3]*q[3];
  float sk = k[0]*k[0] + k[1]*k[1] + k[2]*k[2] + k[3]*k[3];
  sq = waveSum32(sq);
  sk = waveSum32(sk);
  const float rq = rsqrtf(sq * (1.0f / DIM_HEAD) + 1e-5f);
  const float rk = rsqrtf(sk * (1.0f / DIM_HEAD) + 1e-5f);

  float qn[4], kn[4], c[4], s[4];
  #pragma unroll
  for (int i = 0; i < 4; ++i) {
    qn[i] = q[i] * rq * wq[d0 + i];
    kn[i] = k[i] * rk * wk[d0 + i];
    c[i]  = cosb[(size_t)spos * DIM_HEAD + d0 + i];
    s[i]  = sinb[(size_t)spos * DIM_HEAD + d0 + i];
  }
  float qo[4], ko[4];
  #pragma unroll
  for (int p = 0; p < 2; ++p) {          // x_rot[2i]=-x[2i+1], x_rot[2i+1]=x[2i]
    int a = 2 * p, b = 2 * p + 1;
    qo[a] = qn[a] * c[a] - qn[b] * s[a];
    qo[b] = qn[b] * c[b] + qn[a] * s[b];
    ko[a] = kn[a] * c[a] - kn[b] * s[a];
    ko[b] = kn[b] * c[b] + kn[a] * s[b];
  }
  const size_t qkbase = ((size_t)head * SEQ + spos) * DIM_HEAD + d0;
  #pragma unroll
  for (int i = 0; i < 4; ++i) {
    qh[qkbase + i] = f2bf(qo[i]);
    kh[qkbase + i] = f2bf(ko[i]);
    vT[((size_t)head * DIM_HEAD + d0 + i) * SEQ + spos] = f2bf(v[i]);
  }
}

// ---------------------------------------------------------------------------
// SiLU-gated MLP branch -> concat buffer columns [INNER, CAT_DIM)
// ---------------------------------------------------------------------------
__global__ void mlp_silu(const float* __restrict__ hbuf,
                         unsigned short* __restrict__ catb) {
  const long n = (long)SEQ * MLP_HID;
  long i = (long)blockIdx.x * blockDim.x + threadIdx.x;
  long stride = (long)gridDim.x * blockDim.x;
  for (; i < n; i += stride) {
    long sp = i / MLP_HID, j = i % MLP_HID;
    float x1 = hbuf[(size_t)sp * PROJ_OUT + 3 * INNER + j];
    float x2 = hbuf[(size_t)sp * PROJ_OUT + 3 * INNER + MLP_HID + j];
    float y = (x1 / (1.0f + __expf(-x1))) * x2;
    catb[(size_t)sp * CAT_DIM + INNER + j] = f2bf(y);
  }
}

// ---------------------------------------------------------------------------
// Flash attention, one wave per (head, 16-query tile). bf16 WMMA QK^T and PV,
// online softmax across 16-lane row groups, P re-laid out through 1KB LDS.
// ---------------------------------------------------------------------------
__global__ __launch_bounds__(32) void flash_attn(
    const unsigned short* __restrict__ qh, const unsigned short* __restrict__ kh,
    const unsigned short* __restrict__ vT, unsigned short* __restrict__ catb) {
  __shared__ __align__(16) unsigned short P[16 * 32];

  const int head = blockIdx.y, q0 = blockIdx.x * 16;
  const int lane = threadIdx.x, h = lane >> 4, ln = lane & 15;
  const float scale = 0.08838834764831845f;   // 1/sqrt(128)

  // Q A-fragments: 4 chunks of K=32 head dims.
  FragB qf[4];
  const unsigned short* Qrow = qh + ((size_t)head * SEQ + q0 + ln) * DIM_HEAD;
  #pragma unroll
  for (int cchunk = 0; cchunk < 4; ++cchunk) {
    qf[cchunk].q[0] = *(const v4u*)(Qrow + 32 * cchunk + 8 * h);
    qf[cchunk].q[1] = *(const v4u*)(Qrow + 32 * cchunk + 16 + 8 * h);
  }

  v8f o[8] = {};
  float m[8], l[8];
  #pragma unroll
  for (int r = 0; r < 8; ++r) { m[r] = -1e30f; l[r] = 0.0f; }

  const v4u* P4 = (const v4u*)P;

  for (int j0 = 0; j0 < SEQ; j0 += 32) {
    // scores for 32 keys = two 16x16 D fragments
    v8f s0 = {}, s1 = {};
    #pragma unroll
    for (int cchunk = 0; cchunk < 4; ++cchunk) {
      FragB bk;
      const unsigned short* K0 =
          kh + ((size_t)head * SEQ + j0 + ln) * DIM_HEAD + 32 * cchunk + 8 * h;
      bk.q[0] = *(const v4u*)(K0);
      bk.q[1] = *(const v4u*)(K0 + 16);
      s0 = __builtin_amdgcn_wmma_f32_16x16x32_bf16(
          false, qf[cchunk].v, false, bk.v, (short)0, s0, false, false);
      const unsigned short* K1 =
          kh + ((size_t)head * SEQ + j0 + 16 + ln) * DIM_HEAD + 32 * cchunk + 8 * h;
      bk.q[0] = *(const v4u*)(K1);
      bk.q[1] = *(const v4u*)(K1 + 16);
      s1 = __builtin_amdgcn_wmma_f32_16x16x32_bf16(
          false, qf[cchunk].v, false, bk.v, (short)0, s1, false, false);
    }

    // online softmax: row M=r+8h lives across the 16 lanes of this half-wave
    float p0[8], p1[8];
    #pragma unroll
    for (int r = 0; r < 8; ++r) {
      float a = s0[r] * scale, b = s1[r] * scale;
      float nm = fmaxf(m[r], grpMax16(fmaxf(a, b)));
      float alpha = __expf(m[r] - nm);
      m[r] = nm;
      p0[r] = __expf(a - nm);
      p1[r] = __expf(b - nm);
      l[r] = l[r] * alpha + grpSum16(p0[r] + p1[r]);
      #pragma unroll
      for (int nch = 0; nch < 8; ++nch) o[nch][r] *= alpha;
    }

    // D-layout -> A-layout for P via LDS (16 rows x 32 keys, bf16)
    #pragma unroll
    for (int r = 0; r < 8; ++r) {
      P[(r + 8 * h) * 32 + ln]      = f2bf(p0[r]);
      P[(r + 8 * h) * 32 + 16 + ln] = f2bf(p1[r]);
    }
    __syncthreads();
    FragB pf;
    pf.q[0] = P4[ln * 4 + h];
    pf.q[1] = P4[ln * 4 + 2 + h];

    // PV: B = 32 keys x 16 dims per chunk; vT[head][dim][S] is contiguous in key
    #pragma unroll
    for (int nch = 0; nch < 8; ++nch) {
      FragB bv;
      const unsigned short* Vr =
          vT + ((size_t)head * DIM_HEAD + 16 * nch + ln) * SEQ + j0 + 8 * h;
      bv.q[0] = *(const v4u*)(Vr);
      bv.q[1] = *(const v4u*)(Vr + 16);
      o[nch] = __builtin_amdgcn_wmma_f32_16x16x32_bf16(
          false, pf.v, false, bv.v, (short)0, o[nch], false, false);
    }
    __syncthreads();
  }

  // write normalized output into concat buffer columns [0, INNER)
  #pragma unroll
  for (int nch = 0; nch < 8; ++nch)
    #pragma unroll
    for (int r = 0; r < 8; ++r) {
      int row = q0 + r + 8 * h;
      int col = head * DIM_HEAD + 16 * nch + ln;
      catb[(size_t)row * CAT_DIM + col] = f2bf(o[nch][r] / l[r]);
    }
}

// ---------------------------------------------------------------------------
// Host-side orchestration
// ---------------------------------------------------------------------------
extern "C" void kernel_launch(void* const* d_in, const int* in_sizes, int n_in,
                              void* d_out, int out_size, void* d_ws, size_t ws_size,
                              hipStream_t stream) {
  (void)in_sizes; (void)n_in; (void)out_size; (void)ws_size;
  const float* hidden = (const float*)d_in[0];
  const float* cosb   = (const float*)d_in[1];
  const float* sinb   = (const float*)d_in[2];
  const float* Wqkv   = (const float*)d_in[3];
  const float* Wout   = (const float*)d_in[4];
  const float* bout   = (const float*)d_in[5];
  const float* nqw    = (const float*)d_in[6];
  const float* nkw    = (const float*)d_in[7];
  float* out = (float*)d_out;

  char* w = (char*)d_ws;
  size_t off = 0;
  unsigned short* hs_bf = (unsigned short*)(w + off); off += (size_t)SEQ * QUERY_DIM * 2;
  unsigned short* Wq_bf = (unsigned short*)(w + off); off += (size_t)PROJ_OUT * QUERY_DIM * 2;
  unsigned short* Wo_bf = (unsigned short*)(w + off); off += (size_t)QUERY_DIM * CAT_DIM * 2;
  float*          hbuf  = (float*)(w + off);          off += (size_t)SEQ * PROJ_OUT * 4;
  unsigned short* qh    = (unsigned short*)(w + off); off += (size_t)HEADS * SEQ * DIM_HEAD * 2;
  unsigned short* kh    = (unsigned short*)(w + off); off += (size_t)HEADS * SEQ * DIM_HEAD * 2;
  unsigned short* vT    = (unsigned short*)(w + off); off += (size_t)HEADS * SEQ * DIM_HEAD * 2;
  unsigned short* catb  = (unsigned short*)(w + off); off += (size_t)SEQ * CAT_DIM * 2;

  cast_f32_bf16<<<2048, 256, 0, stream>>>(hidden, hs_bf, (long)SEQ * QUERY_DIM);
  cast_f32_bf16<<<8192, 256, 0, stream>>>(Wqkv,   Wq_bf, (long)PROJ_OUT * QUERY_DIM);
  cast_f32_bf16<<<8192, 256, 0, stream>>>(Wout,   Wo_bf, (long)QUERY_DIM * CAT_DIM);

  // h = hidden @ W_qkv_mlp^T : [2048 x 33792]
  gemm_bf16_nt<<<dim3(PROJ_OUT / 128, SEQ / 128), 256, 0, stream>>>(
      hs_bf, Wq_bf, hbuf, nullptr, SEQ, PROJ_OUT, QUERY_DIM);

  qkv_norm_rope<<<dim3(SEQ, HEADS), 32, 0, stream>>>(
      hbuf, cosb, sinb, nqw, nkw, qh, kh, vT);

  mlp_silu<<<8192, 256, 0, stream>>>(hbuf, catb);

  flash_attn<<<dim3(SEQ / 16, HEADS), 32, 0, stream>>>(qh, kh, vT, catb);

  // out = cat @ W_out^T + b_out : [2048 x 3072]
  gemm_bf16_nt<<<dim3(QUERY_DIM / 128, SEQ / 128), 256, 0, stream>>>(
      catb, Wo_bf, out, bout, SEQ, QUERY_DIM, CAT_DIM);
}
